// SinkhornDivergence_7808250544442
// MI455X (gfx1250) — matrix operand order repaired and checked by reading, not compiled, not run on hardware
//
#include <hip/hip_runtime.h>
#include <hip/hip_fp16.h>

typedef _Float16 h8  __attribute__((ext_vector_type(8)));
typedef __attribute__((ext_vector_type(16))) _Float16 v16h;
typedef __attribute__((ext_vector_type(8)))  float    v8f;

#define NN 8192
#define DD 128
#define EPSILON  0.1f
#define EPS_STAB 1e-6f
#define N_ITERS  50
#define ROWCHUNK 128            // rows per col_update partial block

// ---------------- small prep kernels ----------------

__global__ void cvt_f16_kernel(const float* __restrict__ x, _Float16* __restrict__ xh, int n) {
    int i = blockIdx.x * blockDim.x + threadIdx.x;
    if (i < n) xh[i] = (_Float16)x[i];
}

__global__ void rownorm_kernel(const float* __restrict__ x, float* __restrict__ x2, int n) {
    int i = blockIdx.x * blockDim.x + threadIdx.x;
    if (i >= n) return;
    const float4* r = (const float4*)(x + (size_t)i * DD);
    float acc = 0.0f;
#pragma unroll
    for (int t = 0; t < DD / 4; ++t) {
        float4 q = r[t];
        acc += q.x * q.x + q.y * q.y + q.z * q.z + q.w * q.w;
    }
    x2[i] = acc;
}

__global__ void pair_init_kernel(float* __restrict__ u, float* __restrict__ v,
                                 float* __restrict__ vacc, float* __restrict__ ot) {
    int i = blockIdx.x * blockDim.x + threadIdx.x;
    if (i < NN) { u[i] = 1.0f; v[i] = 1.0f; vacc[i] = 0.0f; }
    if (i == 0) *ot = 0.0f;
}

// ---------------- WMMA GEMM + exp -> f16 K ----------------
// Block computes a 128x128 tile of K = exp(-max(x2+y2-2 X.Y^T,0)/eps).
// 8 waves in a 2x4 grid, each wave owns 64x32 = 4x2 subtiles of 16x16,
// accumulated with v_wmma_f32_16x16x32_f16 from LDS-staged f16 panels.

__global__ void __launch_bounds__(256) gemm_exp_kernel(
    const _Float16* __restrict__ A, const _Float16* __restrict__ B,
    const float* __restrict__ x2, const float* __restrict__ y2,
    _Float16* __restrict__ K)
{
    __shared__ _Float16 As[128][72];   // 64 K-values per chunk + 8 pad (144B stride)
    __shared__ _Float16 Bs[128][72];

    const int tid  = threadIdx.x;
    const int lane = tid & 31;
    const int wave = tid >> 5;
    const int r    = lane & 15;        // row-in-subtile (A) / column (B/C)
    const int h    = lane >> 4;        // half-wave selector
    const int wm   = wave >> 2;        // 0..1
    const int wn   = wave & 3;         // 0..3
    const int i0   = blockIdx.y * 128;
    const int j0   = blockIdx.x * 128;

    v8f acc[4][2];
#pragma unroll
    for (int ms = 0; ms < 4; ++ms)
#pragma unroll
        for (int ns = 0; ns < 2; ++ns)
#pragma unroll
            for (int q = 0; q < 8; ++q) acc[ms][ns][q] = 0.0f;

    for (int ko = 0; ko < DD; ko += 64) {
        __syncthreads();
        for (int idx = tid; idx < 128 * 8; idx += 256) {
            int row = idx >> 3, ch = idx & 7;
            *(h8*)&As[row][ch * 8] = *(const h8*)(A + (size_t)(i0 + row) * DD + ko + ch * 8);
            *(h8*)&Bs[row][ch * 8] = *(const h8*)(B + (size_t)(j0 + row) * DD + ko + ch * 8);
        }
        __syncthreads();
#pragma unroll
        for (int kc = 0; kc < 64; kc += 32) {
            v16h af[4], bf[2];
            // A fragment: lanes 0-15 hold K {2t,2t+1 | 16+..}; lanes 16-31 offset +8/+24
#pragma unroll
            for (int ms = 0; ms < 4; ++ms) {
                int row = wm * 64 + ms * 16 + r;
#pragma unroll
                for (int t = 0; t < 8; ++t) {
                    int kb = kc + ((t < 4) ? ((h ? 8 : 0) + 2 * t)
                                           : ((h ? 24 : 16) + 2 * (t - 4)));
                    af[ms][2 * t]     = As[row][kb];
                    af[ms][2 * t + 1] = As[row][kb + 1];
                }
            }
            // B fragment: lane = column N; same K order as A (lanes 0-15 K {0..7,16..23}, lanes 16-31 +8)
#pragma unroll
            for (int ns = 0; ns < 2; ++ns) {
                int row = wn * 32 + ns * 16 + r;
#pragma unroll
                for (int t = 0; t < 8; ++t) {
                    int kb = kc + ((t < 4) ? ((h ? 8 : 0) + 2 * t)
                                           : ((h ? 24 : 16) + 2 * (t - 4)));
                    bf[ns][2 * t]     = Bs[row][kb];
                    bf[ns][2 * t + 1] = Bs[row][kb + 1];
                }
            }
#pragma unroll
            for (int ms = 0; ms < 4; ++ms)
#pragma unroll
                for (int ns = 0; ns < 2; ++ns)
                    acc[ms][ns] = __builtin_amdgcn_wmma_f32_16x16x32_f16(
                        false, af[ms], false, bf[ns], (short)0, acc[ms][ns], false, false);
        }
    }

#pragma unroll
    for (int ms = 0; ms < 4; ++ms) {
        int gi_base = i0 + wm * 64 + ms * 16 + 8 * h;   // C layout: VGPR c -> M = c + 8*h
#pragma unroll
        for (int ns = 0; ns < 2; ++ns) {
            int gj = j0 + wn * 32 + ns * 16 + r;
            float yv = y2[gj];
#pragma unroll
            for (int c = 0; c < 8; ++c) {
                int gi = gi_base + c;
                float s   = acc[ms][ns][c];
                float Cij = fmaxf(x2[gi] + yv - 2.0f * s, 0.0f);
                float kv  = __expf(-Cij * (1.0f / EPSILON));
                K[(size_t)gi * NN + gj] = (_Float16)kv;
            }
        }
    }
}

// ---------------- u update: one wave per row of K ----------------
// u[i] = where(|(K@v)[i]+eps| < eps, fill, fill / max((K@v)[i]+eps, eps))

__global__ void __launch_bounds__(256) row_update_kernel(
    const _Float16* __restrict__ Kmat, const float* __restrict__ vin,
    float* __restrict__ uout, float fill)
{
    __shared__ float vs[NN];
    for (int j = threadIdx.x; j < NN / 4; j += blockDim.x)
        ((float4*)vs)[j] = ((const float4*)vin)[j];
    __syncthreads();

    const int lane = threadIdx.x & 31;
    const int wave = threadIdx.x >> 5;
    const int row  = blockIdx.x * 8 + wave;
    const _Float16* kr = Kmat + (size_t)row * NN;

    float acc = 0.0f;
    for (int j0 = lane * 8; j0 < NN; j0 += 256) {        // 32 iters of 16B loads/lane
        __builtin_prefetch((const void*)(kr + j0 + 2048), 0, 1);  // global_prefetch_b8
        h8 kv = *(const h8*)(kr + j0);
#pragma unroll
        for (int t = 0; t < 8; ++t) acc += (float)kv[t] * vs[j0 + t];
    }
#pragma unroll
    for (int off = 16; off > 0; off >>= 1) acc += __shfl_down(acc, off, 32);
    if (lane == 0) {
        float Kv = acc + EPS_STAB;
        uout[row] = (fabsf(Kv) < EPS_STAB) ? fill : fill / fmaxf(Kv, EPS_STAB);
    }
}

// ---------------- v update part 1: K^T @ u over row-major K ----------------
// Block owns 2048 consecutive columns (8 per thread, 16B loads), streams a
// 128-row chunk, accumulates partials into vacc via f32 L2 atomics.

__global__ void __launch_bounds__(256) col_partial_kernel(
    const _Float16* __restrict__ Kmat, const float* __restrict__ uin,
    float* __restrict__ vacc)
{
    __shared__ float us[ROWCHUNK];
    const int j0 = blockIdx.x * 2048 + threadIdx.x * 8;
    const int i0 = blockIdx.y * ROWCHUNK;
    for (int i = threadIdx.x; i < ROWCHUNK; i += 256) us[i] = uin[i0 + i];
    __syncthreads();

    float acc[8];
#pragma unroll
    for (int t = 0; t < 8; ++t) acc[t] = 0.0f;

    const _Float16* kp = Kmat + (size_t)i0 * NN + j0;
    for (int i = 0; i < ROWCHUNK; ++i) {
        h8 kv = *(const h8*)kp;
        float uv = us[i];
#pragma unroll
        for (int t = 0; t < 8; ++t) acc[t] += (float)kv[t] * uv;
        kp += NN;
    }
#pragma unroll
    for (int t = 0; t < 8; ++t) atomicAdd(&vacc[j0 + t], acc[t]);
}

// ---------------- v update part 2: scale + reset accumulator ----------------

__global__ void scale_update_kernel(float* __restrict__ vacc, float* __restrict__ vout,
                                    float fill) {
    int j = blockIdx.x * blockDim.x + threadIdx.x;
    float Kv = vacc[j] + EPS_STAB;
    vout[j] = (fabsf(Kv) < EPS_STAB) ? fill : fill / fmaxf(Kv, EPS_STAB);
    vacc[j] = 0.0f;                       // ready for next iteration
}

// ---------------- final transport cost: sum_ij u_i K_ij C_ij v_j ----------------

__global__ void __launch_bounds__(256) final_dot_kernel(
    const _Float16* __restrict__ Kmat, const float* __restrict__ u,
    const float* __restrict__ v, float* __restrict__ ot)
{
    __shared__ float vs[NN];
    __shared__ float wsum[8];
    for (int j = threadIdx.x; j < NN / 4; j += blockDim.x)
        ((float4*)vs)[j] = ((const float4*)v)[j];
    __syncthreads();

    const int lane = threadIdx.x & 31;
    const int wave = threadIdx.x >> 5;
    const int row  = blockIdx.x * 8 + wave;
    const _Float16* kr = Kmat + (size_t)row * NN;

    float acc = 0.0f;
    for (int j0 = lane * 8; j0 < NN; j0 += 256) {
        h8 kv = *(const h8*)(kr + j0);
#pragma unroll
        for (int t = 0; t < 8; ++t) {
            float kf = (float)kv[t];
            if (kf > 0.0f) {                       // k->0 limit of k*(-eps ln k) is 0
                float Cij = -EPSILON * __logf(kf); // recover C from stored K
                acc += kf * Cij * vs[j0 + t];
            }
        }
    }
#pragma unroll
    for (int off = 16; off > 0; off >>= 1) acc += __shfl_down(acc, off, 32);
    if (lane == 0) wsum[wave] = acc * u[row];
    __syncthreads();
    if (threadIdx.x == 0) {
        float s = 0.0f;
#pragma unroll
        for (int w = 0; w < 8; ++w) s += wsum[w];
        atomicAdd(ot, s);
    }
}

__global__ void combine_kernel(const float* __restrict__ ot, float* __restrict__ out) {
    float d = ot[0] - 0.5f * (ot[1] + ot[2]);
    if (!isfinite(d)) d = 0.0f;
    out[0] = fmaxf(d, 0.0f);
}

// ---------------- launcher ----------------

extern "C" void kernel_launch(void* const* d_in, const int* in_sizes, int n_in,
                              void* d_out, int out_size, void* d_ws, size_t ws_size,
                              hipStream_t stream)
{
    const float* X = (const float*)d_in[0];
    const float* Y = (const float*)d_in[1];
    float* out = (float*)d_out;

    char* ws = (char*)d_ws;
    size_t off = 0;
    auto alloc = [&](size_t bytes) -> void* {
        void* p = ws + off;
        off += (bytes + 255) & ~(size_t)255;
        return p;
    };
    const size_t kb = (size_t)NN * NN * sizeof(_Float16);
    _Float16* Kxy = (_Float16*)alloc(kb);     // each K (128MB) fits in 192MB L2
    _Float16* Kxx = (_Float16*)alloc(kb);
    _Float16* Kyy = (_Float16*)alloc(kb);
    _Float16* Xh  = (_Float16*)alloc((size_t)NN * DD * sizeof(_Float16));
    _Float16* Yh  = (_Float16*)alloc((size_t)NN * DD * sizeof(_Float16));
    float* x2   = (float*)alloc(NN * sizeof(float));
    float* y2   = (float*)alloc(NN * sizeof(float));
    float* u    = (float*)alloc(NN * sizeof(float));
    float* v    = (float*)alloc(NN * sizeof(float));
    float* vacc = (float*)alloc(NN * sizeof(float));
    float* ot   = (float*)alloc(3 * sizeof(float));

    const int nel = NN * DD;
    cvt_f16_kernel<<<(nel + 255) / 256, 256, 0, stream>>>(X, Xh, nel);
    cvt_f16_kernel<<<(nel + 255) / 256, 256, 0, stream>>>(Y, Yh, nel);
    rownorm_kernel<<<NN / 256, 256, 0, stream>>>(X, x2, NN);
    rownorm_kernel<<<NN / 256, 256, 0, stream>>>(Y, y2, NN);

    dim3 gg(NN / 128, NN / 128);
    gemm_exp_kernel<<<gg, 256, 0, stream>>>(Xh, Yh, x2, y2, Kxy);
    gemm_exp_kernel<<<gg, 256, 0, stream>>>(Xh, Xh, x2, x2, Kxx);
    gemm_exp_kernel<<<gg, 256, 0, stream>>>(Yh, Yh, y2, y2, Kyy);

    const float fill = 1.0f / (float)NN;          // a == b == 1/n
    const _Float16* Ks[3] = {Kxy, Kxx, Kyy};
    dim3 gc(NN / 2048, NN / ROWCHUNK);            // 4 x 64 = 256 blocks

    for (int p = 0; p < 3; ++p) {
        pair_init_kernel<<<NN / 256, 256, 0, stream>>>(u, v, vacc, ot + p);
        for (int it = 0; it < N_ITERS; ++it) {
            row_update_kernel<<<NN / 8, 256, 0, stream>>>(Ks[p], v, u, fill);
            col_partial_kernel<<<gc, 256, 0, stream>>>(Ks[p], u, vacc);
            scale_update_kernel<<<NN / 256, 256, 0, stream>>>(vacc, v, fill);
        }
        final_dot_kernel<<<NN / 8, 256, 0, stream>>>(Ks[p], u, v, ot + p);
    }
    combine_kernel<<<1, 1, 0, stream>>>(ot, out);
}
